// Share_lstm_cell_25305947308035
// MI455X (gfx1250) — matrix operand order repaired and checked by reading, not compiled, not run on hardware
//
#include <hip/hip_runtime.h>

// Share-LSTM cell, fused, MI455X (gfx1250, wave32) — bf16x3 split-precision GEMMs.
//
//   tmp  = input    @ W_ih^T + b_ih   (65536 x 1536, K=256)
//   tmp2 = h_minors @ W_hh^T + b_hh   (65536 x 1024, K=256)
//   tmp3 = s[idx]   @ W_sh^T + b_sh   (65536 x  512, K=256)
//   + LSTM gating epilogue -> h, c_t, s0, s1 (each 65536 x 256, fp32)
//
// fp32 operands are split x = hi + lo (two bf16); A*B ~= Ahi*Bhi + Ahi*Blo + Alo*Bhi
// accumulated in f32 via V_WMMA_F32_16X16X32_BF16 (K=32/instr vs K=4 for the f32 WMMA).
// Error ~2^-16 relative (dropped lo*lo term) — fp32-class accuracy.

typedef float  v2f  __attribute__((ext_vector_type(2)));
typedef float  v4f  __attribute__((ext_vector_type(4)));
typedef float  v8f  __attribute__((ext_vector_type(8)));
typedef __bf16 v4bf __attribute__((ext_vector_type(4)));
typedef __bf16 v8bf __attribute__((ext_vector_type(8)));
typedef __bf16 v16bf __attribute__((ext_vector_type(16)));

#define B_ROWS   65536
#define DH       256
#define M_TILE   16
#define BLOCK    512            // 16 wave32s
#define AS       264            // LDS row stride in bf16 elems (16B pad -> bank spread)

#define WIH_ELEMS (1536 * 256)  // 393216
#define WHH_ELEMS (1024 * 256)  // 262144
#define WSH_ELEMS (512  * 256)  // 131072
#define WHH_OFF   WIH_ELEMS
#define WSH_OFF   (WIH_ELEMS + WHH_ELEMS)
#define WTOT      (WIH_ELEMS + WHH_ELEMS + WSH_ELEMS)  // 786432

__device__ __forceinline__ float fast_sigmoid(float x) {
  return 1.0f / (1.0f + __expf(-x));
}

__device__ __forceinline__ float fast_tanh(float x) {
#if __has_builtin(__builtin_amdgcn_tanhf)
  return __builtin_amdgcn_tanhf(x);       // V_TANH_F32 (CDNA5 trans op)
#else
  return 1.0f - 2.0f / (__expf(2.0f * x) + 1.0f);
#endif
}

// ---------------------------------------------------------------------------
// Pre-pass: split all fp32 weights into bf16 hi/lo arrays in workspace.
// hi[t] = bf16(w[t]); lo[t] = bf16(w[t] - float(hi[t])). 3 MB total, L2-resident.
// ---------------------------------------------------------------------------
__global__ __launch_bounds__(256)
void split_weights_kernel(const float* __restrict__ w_ih,
                          const float* __restrict__ w_hh,
                          const float* __restrict__ w_sh,
                          __bf16* __restrict__ hi, __bf16* __restrict__ lo) {
  const int i = (blockIdx.x * 256 + threadIdx.x) * 4;   // region bounds are %4==0
  if (i >= WTOT) return;
  const float* src;
  int local;
  if (i < WHH_OFF)      { src = w_ih; local = i; }
  else if (i < WSH_OFF) { src = w_hh; local = i - WHH_OFF; }
  else                  { src = w_sh; local = i - WSH_OFF; }
  const v4f v = *(const v4f*)(src + local);
  v4bf h, l;
#pragma unroll
  for (int j = 0; j < 4; ++j) {
    const __bf16 hj = (__bf16)v[j];
    h[j] = hj;
    l[j] = (__bf16)(v[j] - (float)hj);
  }
  *(v4bf*)(hi + i) = h;
  *(v4bf*)(lo + i) = l;
}

// A-fragment (bf16 16x32): lane holds two contiguous 16B
// chunks at K = 16*half and K = 16*half + 8 relative to k0.
__device__ __forceinline__ v16bf load_a_frag(const __bf16* base, int k0) {
  const v8bf c1 = *(const v8bf*)(base + k0);
  const v8bf c2 = *(const v8bf*)(base + k0 + 8);
  return __builtin_shufflevector(c1, c2, 0, 1, 2, 3, 4, 5, 6, 7,
                                 8, 9, 10, 11, 12, 13, 14, 15);
}

#define WMMA_BF16(A, Bm, C) \
  __builtin_amdgcn_wmma_f32_16x16x32_bf16(false, (A), false, (Bm), (short)0, (C), false, false)

__global__ __launch_bounds__(BLOCK, 1)
void share_lstm_cell_kernel(
    const float* __restrict__ input,     // (B, 256)
    const float* __restrict__ h_minors,  // (B, 256)
    const float* __restrict__ c_minors,  // (B, 256)
    const float* __restrict__ s_minors,  // (2, B, 256)
    const float* __restrict__ mask,      // (B,)
    const __bf16* __restrict__ whi,      // split weights, ih|hh|sh concatenated
    const __bf16* __restrict__ wlo,
    const float* __restrict__ b_ih, const float* __restrict__ b_hh,
    const float* __restrict__ b_sh,
    const int*   __restrict__ idx_p,
    float* __restrict__ out)             // h | c_t | s0 | s1, each (B, 256)
{
  // LDS: 3 sources x {hi,lo} x 16 rows x 264 bf16  = 50688 B
  __shared__ __bf16 ldsA[3 * 2 * M_TILE * AS];

  const int tid = threadIdx.x;
  const int m0  = blockIdx.x * M_TILE;

  const int    idx   = idx_p[0];
  const float* s_sel = s_minors + (size_t)idx * (size_t)B_ROWS * DH;

  // ------- Stage activation tiles into LDS, splitting fp32 -> bf16 hi/lo ----
  {
    const float* s0p = input    + (size_t)m0 * DH;
    const float* s1p = h_minors + (size_t)m0 * DH;
    const float* s2p = s_sel    + (size_t)m0 * DH;
    for (int t = tid; t < 3 * M_TILE * (DH / 4); t += BLOCK) {  // 3072 f32x4 chunks
      const int s   = t >> 10;
      const int rem = t & 1023;
      const int r   = rem >> 6;
      const int c4  = rem & 63;
      const float* sp = (s == 0) ? s0p : (s == 1) ? s1p : s2p;
      const v4f v = *(const v4f*)(sp + r * DH + c4 * 4);
      v4bf h, l;
#pragma unroll
      for (int j = 0; j < 4; ++j) {
        const __bf16 hj = (__bf16)v[j];
        h[j] = hj;
        l[j] = (__bf16)(v[j] - (float)hj);
      }
      *(v4bf*)&ldsA[(s * 2 + 0) * M_TILE * AS + r * AS + c4 * 4] = h;
      *(v4bf*)&ldsA[(s * 2 + 1) * M_TILE * AS + r * AS + c4 * 4] = l;
    }
  }
  __syncthreads();

  // ------- Per-wave WMMA setup ---------------------------------------------
  const int lane = tid & 31;
  const int wave = tid >> 5;
  const int n0   = wave * 16;       // this wave's 16-column strip of H
  const int nl   = lane & 15;       // M (A frag) / N (B frag) within tile
  const int h    = lane >> 4;       // half-wave selector
  const int n    = n0 + nl;         // absolute output column in [0, 256)

  // A-fragment lane base: row = nl, first chunk at K = 16*h
  const __bf16* aInHiP = &ldsA[0 * M_TILE * AS + nl * AS + 16 * h];
  const __bf16* aInLoP = &ldsA[1 * M_TILE * AS + nl * AS + 16 * h];
  const __bf16* aHHiP  = &ldsA[2 * M_TILE * AS + nl * AS + 16 * h];
  const __bf16* aHLoP  = &ldsA[3 * M_TILE * AS + nl * AS + 16 * h];
  const __bf16* aSHiP  = &ldsA[4 * M_TILE * AS + nl * AS + 16 * h];
  const __bf16* aSLoP  = &ldsA[5 * M_TILE * AS + nl * AS + 16 * h];

  // B fragments: B[k][n] = W[n][k]; lane reads 16 contiguous bf16 (32 B)
  // of weight row (gate*256 + n) at K = k0 + 16*h.
  int off[12];
#pragma unroll
  for (int g = 0; g < 6; ++g) off[g]      = (g * DH + n) * DH + 16 * h;            // ih
#pragma unroll
  for (int g = 0; g < 4; ++g) off[6 + g]  = WHH_OFF + (g * DH + n) * DH + 16 * h;  // hh
#pragma unroll
  for (int g = 0; g < 2; ++g) off[10 + g] = WSH_OFF + (g * DH + n) * DH + 16 * h;  // sh

#pragma unroll
  for (int g = 0; g < 12; ++g) {
    __builtin_prefetch(whi + off[g], 0, 0);   // global_prefetch_b8
    __builtin_prefetch(wlo + off[g], 0, 0);
  }

  v8f acc[12] = {};   // 0..5 = ih(i,f,o,chat,s0',s1'); 6..9 = hh; 10..11 = sh

  // ------- K loop: 8 steps x (12 gates x 3 bf16 WMMA) ----------------------
#pragma unroll 2
  for (int k0 = 0; k0 < DH; k0 += 32) {
    const v16bf aInHi = load_a_frag(aInHiP, k0);
    const v16bf aInLo = load_a_frag(aInLoP, k0);
    const v16bf aHHi  = load_a_frag(aHHiP, k0);
    const v16bf aHLo  = load_a_frag(aHLoP, k0);
    const v16bf aSHi  = load_a_frag(aSHiP, k0);
    const v16bf aSLo  = load_a_frag(aSLoP, k0);
#pragma unroll
    for (int g = 0; g < 6; ++g) {
      const v16bf bHi = *(const v16bf*)(whi + off[g] + k0);
      const v16bf bLo = *(const v16bf*)(wlo + off[g] + k0);
      acc[g] = WMMA_BF16(aInHi, bHi, acc[g]);
      acc[g] = WMMA_BF16(aInHi, bLo, acc[g]);
      acc[g] = WMMA_BF16(aInLo, bHi, acc[g]);
    }
#pragma unroll
    for (int g = 6; g < 10; ++g) {
      const v16bf bHi = *(const v16bf*)(whi + off[g] + k0);
      const v16bf bLo = *(const v16bf*)(wlo + off[g] + k0);
      acc[g] = WMMA_BF16(aHHi, bHi, acc[g]);
      acc[g] = WMMA_BF16(aHHi, bLo, acc[g]);
      acc[g] = WMMA_BF16(aHLo, bHi, acc[g]);
    }
#pragma unroll
    for (int g = 10; g < 12; ++g) {
      const v16bf bHi = *(const v16bf*)(whi + off[g] + k0);
      const v16bf bLo = *(const v16bf*)(wlo + off[g] + k0);
      acc[g] = WMMA_BF16(aSHi, bHi, acc[g]);
      acc[g] = WMMA_BF16(aSHi, bLo, acc[g]);
      acc[g] = WMMA_BF16(aSLo, bHi, acc[g]);
    }
  }

  // ------- Epilogue (all in registers) -------------------------------------
  const float bi  = b_ih[0 * DH + n] + b_hh[0 * DH + n];
  const float bf  = b_ih[1 * DH + n] + b_hh[1 * DH + n];
  const float bo  = b_ih[2 * DH + n] + b_hh[2 * DH + n];
  const float bc  = b_ih[3 * DH + n] + b_hh[3 * DH + n];
  const float bs0 = b_ih[4 * DH + n] + b_sh[0 * DH + n];
  const float bs1 = b_ih[5 * DH + n] + b_sh[1 * DH + n];

  float* __restrict__ out_h  = out;
  float* __restrict__ out_c  = out + 1ull * B_ROWS * DH;
  float* __restrict__ out_s0 = out + 2ull * B_ROWS * DH;
  float* __restrict__ out_s1 = out + 3ull * B_ROWS * DH;

  // C/D layout: vgpr e, lane -> row = m0 + e + 8*h, col = n
#pragma unroll
  for (int e = 0; e < 8; ++e) {
    const int    row = m0 + e + 8 * h;
    const size_t o   = (size_t)row * DH + n;
    const float  mrow = mask[row];
    const float  cm   = c_minors[o];

    const float ig  = fast_sigmoid(acc[0][e] + acc[6][e]  + bi);
    const float fg  = fast_sigmoid(acc[1][e] + acc[7][e]  + bf);
    const float og  = fast_sigmoid(acc[2][e] + acc[8][e]  + bo);
    const float ch  =              acc[3][e] + acc[9][e]  + bc;
    const float bg0 = fast_sigmoid(acc[4][e] + acc[10][e] + bs0);
    const float bg1 = fast_sigmoid(acc[5][e] + acc[11][e] + bs1);

    const float ct = (fg * cm + ig * fast_tanh(ch)) * mrow;
    const float tc = fast_tanh(ct) * mrow;

    out_h [o] = og * tc;
    out_c [o] = ct;
    out_s0[o] = bg0 * tc;
    out_s1[o] = bg1 * tc;
  }
}

extern "C" void kernel_launch(void* const* d_in, const int* in_sizes, int n_in,
                              void* d_out, int out_size, void* d_ws, size_t ws_size,
                              hipStream_t stream) {
  (void)in_sizes; (void)n_in; (void)out_size; (void)ws_size;

  const float* input    = (const float*)d_in[0];
  const float* h_minors = (const float*)d_in[1];
  const float* c_minors = (const float*)d_in[2];
  const float* s_minors = (const float*)d_in[3];
  const float* mask     = (const float*)d_in[4];
  const float* w_ih     = (const float*)d_in[5];
  const float* w_hh     = (const float*)d_in[6];
  const float* w_sh     = (const float*)d_in[7];
  const float* b_ih     = (const float*)d_in[8];
  const float* b_hh     = (const float*)d_in[9];
  const float* b_sh     = (const float*)d_in[10];
  const int*   idx      = (const int*)d_in[11];
  float*       out      = (float*)d_out;

  __bf16* whi = (__bf16*)d_ws;          // WTOT bf16
  __bf16* wlo = whi + WTOT;             // WTOT bf16  (3 MB total)

  // 1) split weights into bf16 hi/lo (deterministic, every call)
  split_weights_kernel<<<dim3((WTOT / 4 + 255) / 256), dim3(256), 0, stream>>>(
      w_ih, w_hh, w_sh, whi, wlo);

  // 2) fused LSTM cell
  share_lstm_cell_kernel<<<dim3(B_ROWS / M_TILE), dim3(BLOCK), 0, stream>>>(
      input, h_minors, c_minors, s_minors, mask,
      whi, wlo, b_ih, b_hh, b_sh, idx, out);
}
